// GCN_FCN_Type1_2740189135369
// MI455X (gfx1250) — matrix operand order repaired and checked by reading, not compiled, not run on hardware
//
#include <hip/hip_runtime.h>
#include <hip/hip_bf16.h>

typedef __attribute__((ext_vector_type(16))) _Float16 v16h;
typedef __attribute__((ext_vector_type(8)))  float    v8f;

#define NB      32
#define NNODES  50000
#define NEDGE   800000
#define NEDGE4  (NEDGE / 4)             // 200000 (exact)
#define TILES_PER_BATCH (NNODES / 16)   // 3125 (exact)

// ---------------- pass 0: zero scratch ----------------
__global__ void k_zero(float* __restrict__ p, int n) {
  int i = blockIdx.x * blockDim.x + threadIdx.x;
  int stride = gridDim.x * blockDim.x;
  for (; i < n; i += stride) p[i] = 0.0f;
}

// ---------------- pass 1: degree histogram over dst ----------------
// 2-D grid: blockIdx.y = batch. int4 loads: 4 edges per thread.
__global__ void k_degree(const int* __restrict__ ei, float* __restrict__ deg) {
  int b = blockIdx.y;
  int i = blockIdx.x * blockDim.x + threadIdx.x;      // int4 index
  if (i >= NEDGE4) return;
  const int4* dst4 = (const int4*)(ei + (size_t)b * 2 * NEDGE + NEDGE);
  float* dg = deg + (size_t)b * NNODES;
  int4 d = dst4[i];
  atomicAdd(&dg[d.x], 1.0f);
  atomicAdd(&dg[d.y], 1.0f);
  atomicAdd(&dg[d.z], 1.0f);
  atomicAdd(&dg[d.w], 1.0f);
}

// ---------------- pass 2: dinv = rsqrt(deg + 1) in place ----------------
__global__ void k_dinv(float* __restrict__ d) {
  int n = NB * NNODES;
  int i = blockIdx.x * blockDim.x + threadIdx.x;
  int stride = gridDim.x * blockDim.x;
  for (; i < n; i += stride) d[i] = rsqrtf(d[i] + 1.0f);
}

// ---------------- pass 3: s[src] += dinv[dst] ----------------
__global__ void k_srcsum(const int* __restrict__ ei, const float* __restrict__ dinv,
                         float* __restrict__ s) {
  int b = blockIdx.y;
  int i = blockIdx.x * blockDim.x + threadIdx.x;      // int4 index
  if (i >= NEDGE4) return;
  const int* base = ei + (size_t)b * 2 * NEDGE;
  const int4* src4 = (const int4*)base;
  const int4* dst4 = (const int4*)(base + NEDGE);
  const float* dv = dinv + (size_t)b * NNODES;
  float* sb = s + (size_t)b * NNODES;
  int4 sv = src4[i];
  int4 dd = dst4[i];
  atomicAdd(&sb[sv.x], dv[dd.x]);
  atomicAdd(&sb[sv.y], dv[dd.y]);
  atomicAdd(&sb[sv.z], dv[dd.z]);
  atomicAdd(&sb[sv.w], dv[dd.w]);
}

// ---------------- pass 4: h = xg @ W via WMMA, fused weighted pooling ----------------
// One wave = one 16-node tile. pooled_sum[b][f] += sum_n h[n][f] * dinv[n]*(s[n]+dinv[n])
__global__ void k_wmma_pool(const float* __restrict__ gcn_x,
                            const float* __restrict__ Wg,
                            const float* __restrict__ dinv,
                            const float* __restrict__ s,
                            float* __restrict__ pooled) {   // pooled[NB][16]
  int gwave = (int)((blockIdx.x * blockDim.x + threadIdx.x) >> 5);
  int lane  = threadIdx.x & 31;
  if (gwave >= NB * TILES_PER_BATCH) return;   // wave-uniform (grid sized exactly)

  int b         = gwave / TILES_PER_BATCH;
  int node_base = (gwave % TILES_PER_BATCH) * 16;

  // --- A fragment: 16x32 f16, rows = nodes, K = features (10 real, rest 0) ---
  // lanes 0-15: row M=lane, elems 0..7 = K0..7 (elems 8..15 = K16..23 = 0)
  // lanes 16-31: row M=lane-16, elems 0..7 = K8..15 (only K8,K9 nonzero)
  int r = lane & 15;
  const float* xrow = gcn_x + ((size_t)b * NNODES + node_base + r) * 10;
  int kbase = (lane < 16) ? 0 : 8;

  v16h a;
  #pragma unroll
  for (int i = 0; i < 16; ++i) a[i] = (_Float16)0.0f;
  #pragma unroll
  for (int i = 0; i < 8; ++i) {
    int k = kbase + i;
    float v = (k < 10) ? xrow[k] : 0.0f;
    a[i] = (_Float16)v;
  }

  // --- B fragment: 32x16 f16, column N=lane&15 (lanes 0-15 elems 0..7 = K0..7; 16-31 elems 0..7 = K8..15) ---
  v16h bw;
  #pragma unroll
  for (int i = 0; i < 16; ++i) bw[i] = (_Float16)0.0f;
  if (r < 10) {
    #pragma unroll
    for (int i = 0; i < 8; ++i) { int k = kbase + i; bw[i] = (_Float16)((k < 10) ? Wg[k * 10 + r] : 0.0f); }
  }

  v8f c = {};
  c = __builtin_amdgcn_wmma_f32_16x16x32_f16(false, a, false, bw, (short)0, c,
                                             false, false);

  // per-node pooling weight w = dinv*(s+dinv); lane holds node (lane&15)
  size_t nidx = (size_t)b * NNODES + node_base + r;
  float di = dinv[nidx];
  float sv = s[nidx];
  float wv = di * (sv + di);

  // D layout: lane L holds feature N=L&15 for rows kbase..kbase+7 (VGPR j = row kbase+j)
  float acc = 0.0f;
  #pragma unroll
  for (int j = 0; j < 8; ++j) {
    float wj = __shfl(wv, kbase + j, 32);
    acc += c[j] * wj;
  }
  acc += __shfl_down(acc, 16, 32);               // combine rows 0-7 and 8-15
  if (lane < 10) atomicAdd(&pooled[b * 16 + lane], acc);
}

// ---------------- pass 5: tiny MLP + log_softmax over axis 0 (single block) ----------------
__global__ void k_mlp(const float* __restrict__ pooled, const float* __restrict__ x,
                      const float* __restrict__ bg,
                      const float* __restrict__ W1, const float* __restrict__ b1,
                      const float* __restrict__ W2, const float* __restrict__ b2,
                      const float* __restrict__ Wo, const float* __restrict__ bo,
                      float* __restrict__ out) {
  __shared__ float hcat[NB][20];
  __shared__ float h1[NB][80];
  __shared__ float h2[NB][160];
  __shared__ float o[NB][4];
  int t = threadIdx.x, nt = blockDim.x;

  for (int i = t; i < NB * 20; i += nt) {
    int b = i / 20, j = i % 20;
    hcat[b][j] = (j < 10) ? (pooled[b * 16 + j] * (1.0f / NNODES) + bg[j])
                          : x[b * 10 + (j - 10)];
  }
  __syncthreads();
  for (int i = t; i < NB * 80; i += nt) {
    int b = i / 80, j = i % 80;
    float acc = b1[j];
    #pragma unroll
    for (int k = 0; k < 20; ++k) acc += hcat[b][k] * W1[k * 80 + j];
    h1[b][j] = (acc > 0.0f) ? acc : 0.01f * acc;
  }
  __syncthreads();
  for (int i = t; i < NB * 160; i += nt) {
    int b = i / 160, j = i % 160;
    float acc = b2[j];
    for (int k = 0; k < 80; ++k) acc += h1[b][k] * W2[k * 160 + j];
    h2[b][j] = (acc > 0.0f) ? acc : 0.01f * acc;
  }
  __syncthreads();
  for (int i = t; i < NB * 4; i += nt) {
    int b = i / 4, j = i % 4;
    float acc = bo[j];
    for (int k = 0; k < 160; ++k) acc += h2[b][k] * Wo[k * 4 + j];
    o[b][j] = acc;
  }
  __syncthreads();
  if (t < 4) {
    int j = t;
    float m = -1e30f;
    for (int b = 0; b < NB; ++b) m = fmaxf(m, o[b][j]);
    float sum = 0.0f;
    for (int b = 0; b < NB; ++b) sum += expf(o[b][j] - m);
    float lse = m + logf(sum);
    for (int b = 0; b < NB; ++b) out[b * 4 + j] = o[b][j] - lse;
  }
}

extern "C" void kernel_launch(void* const* d_in, const int* in_sizes, int n_in,
                              void* d_out, int out_size, void* d_ws, size_t ws_size,
                              hipStream_t stream) {
  const float* x     = (const float*)d_in[0];
  const float* gcn_x = (const float*)d_in[1];
  const int*   ei    = (const int*)d_in[2];
  const float* Wg    = (const float*)d_in[3];
  const float* bg    = (const float*)d_in[4];
  const float* W1    = (const float*)d_in[5];
  const float* b1    = (const float*)d_in[6];
  const float* W2    = (const float*)d_in[7];
  const float* b2    = (const float*)d_in[8];
  const float* Wo    = (const float*)d_in[9];
  const float* bo    = (const float*)d_in[10];
  float* out = (float*)d_out;

  // workspace: deg/dinv [NB*NNODES] | s [NB*NNODES] | pooled [NB*16]  (~12.8 MB)
  float* deg    = (float*)d_ws;
  float* s      = deg + (size_t)NB * NNODES;
  float* pooled = s + (size_t)NB * NNODES;

  int nzero = NB * NNODES * 2 + NB * 16;         // contiguous zero fill
  k_zero<<<2048, 256, 0, stream>>>(deg, nzero);

  dim3 egrid((NEDGE4 + 255) / 256, NB);          // blockIdx.y = batch, no div/mod
  k_degree<<<egrid, 256, 0, stream>>>(ei, deg);
  k_dinv<<<2048, 256, 0, stream>>>(deg);         // deg -> dinv in place
  k_srcsum<<<egrid, 256, 0, stream>>>(ei, deg, s);

  int nwaves  = NB * TILES_PER_BATCH;            // 100000, exact multiple of 8
  int nblocks = nwaves / 8;                      // 256 threads = 8 waves per block
  k_wmma_pool<<<nblocks, 256, 0, stream>>>(gcn_x, Wg, deg, s, pooled);

  k_mlp<<<1, 128, 0, stream>>>(pooled, x, bg, W1, b1, W2, b2, Wo, bo, out);
}
